// UniTSBlock_75488345194609
// MI455X (gfx1250) — compile-verified, matrix operand order checked by reading, not run on hardware
//
#include <hip/hip_runtime.h>
#include <hip/hip_bf16.h>
#include <math.h>

// ---------------- CDNA5 WMMA types ----------------
typedef __attribute__((ext_vector_type(16))) __bf16 v16bf;
typedef __attribute__((ext_vector_type(8)))  float  v8f;

union Frag16 { v16bf v; unsigned int u[8]; unsigned short h[16]; };
union AccF   { v8f v; float f[8]; };

static constexpr int  Dm   = 512;
static constexpr int  DFFm = 2048;
static constexpr int  Bm = 8, NVm = 16, Sm = 512, Hm = 8, HDm = 64;
static constexpr long Tm = (long)Bm * NVm * Sm;   // 65536 tokens

__device__ inline unsigned short f2bf(float f) {
  union { float f; unsigned int u; } c; c.f = f;
  unsigned int u = c.u + 0x7fffu + ((c.u >> 16) & 1u);   // round-to-nearest-even
  return (unsigned short)(u >> 16);
}

// Load a 16x32 bf16 fragment (A layout; also used for B via symmetric layout).
// Per ISA 7.12.2: lane<16 holds row (lane&15), k={0..7,16..23}; lane>=16 same row, k={8..15,24..31}.
__device__ inline Frag16 frag_ld(const unsigned short* tile, long stride) {
  const int lane = threadIdx.x & 31;
  const unsigned short* p = tile + (long)(lane & 15) * stride + ((lane & 16) ? 8 : 0);
  Frag16 f;
  *(uint4*)&f.u[0] = *(const uint4*)p;
  *(uint4*)&f.u[4] = *(const uint4*)(p + 16);
  return f;
}

__device__ inline v8f wmma_bf16(const Frag16& a, const Frag16& b, v8f c) {
  return __builtin_amdgcn_wmma_f32_16x16x32_bf16(false, a.v, false, b.v, (short)0, c, false, false);
}

// ---- CDNA5 async global->LDS copy (ASYNCcnt tracked, ISA 10.x / 15.18.3) ----
// LDS aperture: generic shared pointer low 32 bits == LDS byte offset (ISA 10.2).
__device__ inline void async_ld_b128(const void* gaddr, void* lds_ptr) {
  unsigned loff = (unsigned)(unsigned long long)(uintptr_t)lds_ptr;
  unsigned long long ga = (unsigned long long)(uintptr_t)gaddr;
  asm volatile("global_load_async_to_lds_b128 %0, %1, off"
               :: "v"(loff), "v"(ga) : "memory");
}
__device__ inline void wait_async0() {
  asm volatile("s_wait_asynccnt 0" ::: "memory");
}

// ---------------- fp32 -> bf16 convert ----------------
__global__ __launch_bounds__(256) void cvt_bf16_k(const float* __restrict__ s,
                                                  unsigned short* __restrict__ d, long n) {
  long i = (long)blockIdx.x * 256 + threadIdx.x;
  long stride = (long)gridDim.x * 256;
  for (; i < n; i += stride) d[i] = f2bf(s[i]);
}

// ---------------- WMMA GEMM: Y[M,N] = X[M,K] @ W[N,K]^T ----------------
// Workgroup tile 256(M) x 64(N); weight k-slab async-DMA'd to LDS, double buffered,
// shared by all 8 waves. Each wave computes 32x64 (8 WMMAs / k-step of 32).
// EPI: 0 = store bf16 ; 1 = +bias bf16 ; 2 = +bias +GELU bf16 ; 3 = +bias fp32
template <int EPI>
__global__ __launch_bounds__(256) void gemm_bf16_k(
    const unsigned short* __restrict__ X, const unsigned short* __restrict__ W,
    const float* __restrict__ bias, float* __restrict__ Yf,
    unsigned short* __restrict__ Yh, int M, int N, int K) {
  const int tid = threadIdx.x;
  const int wid = tid >> 5;
  const int lane = tid & 31;
  const int m0 = blockIdx.x * 256 + wid * 32;
  const int n0 = blockIdx.y * 64;

  __shared__ __align__(16) unsigned short Bl[2][64 * 32];   // [n][k] slab, 2 buffers

  // staging map: thread -> (weight row n, 8-element k-chunk); 256 b128 DMAs = full slab
  const int sn = tid >> 2;          // 0..63
  const int sc = (tid & 3) * 8;     // 0,8,16,24

  AccF acc[2][4];
#pragma unroll
  for (int mm = 0; mm < 2; ++mm)
#pragma unroll
    for (int t = 0; t < 4; ++t)
#pragma unroll
      for (int r = 0; r < 8; ++r) acc[mm][t].f[r] = 0.0f;

  const unsigned short* Wg = W + (long)(n0 + sn) * K + sc;
  async_ld_b128(Wg, &Bl[0][sn * 32 + sc]);                  // preload slab 0

  const int KT = K >> 5;
  for (int kt = 0; kt < KT; ++kt) {
    const int k0 = kt << 5;
    wait_async0();          // own wave's outstanding fill of Bl[kt&1]
    __syncthreads();        // all fills visible; all prior readers of the alt buffer done
    if (kt + 1 < KT)
      async_ld_b128(Wg + k0 + 32, &Bl[(kt + 1) & 1][sn * 32 + sc]);

    if (kt + 1 < KT) {      // prefetch streamed A rows for the next k-step
      const unsigned short* nx =
          X + (long)(m0 + (lane & 15)) * K + k0 + 32 + ((lane & 16) ? 8 : 0);
      __builtin_prefetch(nx, 0, 0);
      __builtin_prefetch(nx + 16 * (long)K, 0, 0);
    }

    Frag16 a0 = frag_ld(X + (long)m0 * K + k0, (long)K);
    Frag16 a1 = frag_ld(X + (long)(m0 + 16) * K + k0, (long)K);
    const unsigned short* cur = Bl[kt & 1];
#pragma unroll
    for (int t = 0; t < 4; ++t) {
      Frag16 b = frag_ld(cur + t * 16 * 32, 32);
      acc[0][t].v = wmma_bf16(a0, b, acc[0][t].v);
      acc[1][t].v = wmma_bf16(a1, b, acc[1][t].v);
    }
  }

  const int rowoff = (lane & 16) ? 8 : 0;
  const int coloff = lane & 15;
#pragma unroll
  for (int t = 0; t < 4; ++t) {
    const int c = n0 + t * 16 + coloff;
    const float bv = (EPI >= 1) ? bias[c] : 0.0f;
#pragma unroll
    for (int mm = 0; mm < 2; ++mm)
#pragma unroll
      for (int r = 0; r < 8; ++r) {
        const long row = m0 + mm * 16 + rowoff + r;
        float v = acc[mm][t].f[r] + bv;
        if (EPI == 2) v = 0.5f * v * (1.0f + erff(v * 0.70710678118654752f));
        if (EPI == 3) Yf[row * (long)N + c] = v;
        else          Yh[row * (long)N + c] = f2bf(v);
      }
  }
}

// ---------------- sequence attention (flash-style), L=512, hd=64 ----------------
__global__ __launch_bounds__(256) void attn_seq_k(
    const unsigned short* __restrict__ Q, const unsigned short* __restrict__ Kb,
    const unsigned short* __restrict__ Vb, unsigned short* __restrict__ O) {
  const int n = blockIdx.x;       // batch (B*NV = 128)
  const int h = blockIdx.y;       // head
  const int qt = blockIdx.z;      // q tile of 128 rows
  const int tid = threadIdx.x;
  const int wid = tid >> 5, lane = tid & 31;
  const int rowoff = (lane & 16) ? 8 : 0;
  const int coloff = lane & 15;
  const long base = (long)n * Sm;
  const int hc = h * HDm;
  const int q0 = qt * 128 + wid * 16;

  __shared__ __align__(16) unsigned short Vt[HDm * 32];      // V^T tile [hd][kv]
  __shared__ __align__(16) unsigned short Pst[8][16 * 32];   // per-wave P staging

  Frag16 qf0 = frag_ld(Q + (base + q0) * Dm + hc,      (long)Dm);
  Frag16 qf1 = frag_ld(Q + (base + q0) * Dm + hc + 32, (long)Dm);

  AccF o[4];
#pragma unroll
  for (int t = 0; t < 4; ++t)
#pragma unroll
    for (int r = 0; r < 8; ++r) o[t].f[r] = 0.0f;
  float m[8], l[8];
#pragma unroll
  for (int r = 0; r < 8; ++r) { m[r] = -1e30f; l[r] = 0.0f; }

  for (int kb = 0; kb < Sm; kb += 32) {
    __syncthreads();
    {   // stage V^T: one b128 load + 8 scattered ds stores per thread
      const int kv = tid >> 3, c8 = (tid & 7) * 8;
      uint4 w = *(const uint4*)(Vb + (base + kb + kv) * Dm + hc + c8);
      const unsigned short* e = (const unsigned short*)&w;
#pragma unroll
      for (int j = 0; j < 8; ++j) Vt[(c8 + j) * 32 + kv] = e[j];
    }
    __syncthreads();

    AccF s0, s1;
#pragma unroll
    for (int r = 0; r < 8; ++r) { s0.f[r] = 0.0f; s1.f[r] = 0.0f; }
    Frag16 k00 = frag_ld(Kb + (base + kb     ) * Dm + hc,      (long)Dm);
    Frag16 k01 = frag_ld(Kb + (base + kb     ) * Dm + hc + 32, (long)Dm);
    Frag16 k10 = frag_ld(Kb + (base + kb + 16) * Dm + hc,      (long)Dm);
    Frag16 k11 = frag_ld(Kb + (base + kb + 16) * Dm + hc + 32, (long)Dm);
    s0.v = wmma_bf16(qf0, k00, s0.v); s0.v = wmma_bf16(qf1, k01, s0.v);
    s1.v = wmma_bf16(qf0, k10, s1.v); s1.v = wmma_bf16(qf1, k11, s1.v);

    unsigned short* Pw = Pst[wid];
#pragma unroll
    for (int r = 0; r < 8; ++r) {
      float a = s0.f[r] * 0.125f;       // 1/sqrt(64)
      float b = s1.f[r] * 0.125f;
      float mx = fmaxf(a, b);
#pragma unroll
      for (int d = 8; d >= 1; d >>= 1) mx = fmaxf(mx, __shfl_xor(mx, d));
      float mn = fmaxf(m[r], mx);
      float p0 = __expf(a - mn), p1 = __expf(b - mn);
      float sum = p0 + p1;
#pragma unroll
      for (int d = 8; d >= 1; d >>= 1) sum += __shfl_xor(sum, d);
      float alpha = __expf(m[r] - mn);
      l[r] = l[r] * alpha + sum;
      m[r] = mn;
#pragma unroll
      for (int t = 0; t < 4; ++t) o[t].f[r] *= alpha;
      Pw[(rowoff + r) * 32 + coloff]      = f2bf(p0);
      Pw[(rowoff + r) * 32 + 16 + coloff] = f2bf(p1);
    }
    Frag16 pf = frag_ld(Pw, 32);        // same-wave LDS ops are in order
#pragma unroll
    for (int t = 0; t < 4; ++t) {
      Frag16 vf = frag_ld(Vt + (t * 16) * 32, 32);
      o[t].v = wmma_bf16(pf, vf, o[t].v);
    }
  }

#pragma unroll
  for (int r = 0; r < 8; ++r) {
    const float inv = 1.0f / l[r];
    const long row = base + q0 + rowoff + r;
#pragma unroll
    for (int t = 0; t < 4; ++t)
      O[row * Dm + hc + t * 16 + coloff] = f2bf(o[t].f[r] * inv);
  }
}

// ---------------- variable attention, L = NV = 16 (one WMMA tile/head) ----------------
__global__ __launch_bounds__(256) void attn_var_k(
    const unsigned short* __restrict__ Q, const unsigned short* __restrict__ Kb,
    const unsigned short* __restrict__ Vb, unsigned short* __restrict__ O) {
  const int bs = blockIdx.x;                 // b*S + s
  const int b = bs >> 9, s = bs & 511;
  const int tid = threadIdx.x;
  const int h = tid >> 5, lane = tid & 31;   // wave == head
  const int rowoff = (lane & 16) ? 8 : 0;
  const int coloff = lane & 15;
  const long row0 = (long)b * (NVm * Sm) + s;    // v = 0 token row
  const long estride = (long)Sm * Dm;            // element stride between v rows
  const int hc = h * HDm;

  __shared__ __align__(16) unsigned short Vt[Dm * NVm];     // V^T [c][v], 16KB
  __shared__ __align__(16) unsigned short Pst[8][16 * 32];

  for (int i = tid; i < NVm * (Dm / 8); i += 256) {   // b128 load + 8 ds scatter stores
    const int v = i >> 6, c8 = (i & 63) * 8;
    uint4 w = *(const uint4*)(Vb + (row0 + (long)v * Sm) * Dm + c8);
    const unsigned short* e = (const unsigned short*)&w;
#pragma unroll
    for (int j = 0; j < 8; ++j) Vt[(c8 + j) * NVm + v] = e[j];
  }
  __syncthreads();

  Frag16 qf0 = frag_ld(Q  + row0 * Dm + hc,      estride);
  Frag16 qf1 = frag_ld(Q  + row0 * Dm + hc + 32, estride);
  Frag16 kf0 = frag_ld(Kb + row0 * Dm + hc,      estride);
  Frag16 kf1 = frag_ld(Kb + row0 * Dm + hc + 32, estride);

  AccF s0;
#pragma unroll
  for (int r = 0; r < 8; ++r) s0.f[r] = 0.0f;
  s0.v = wmma_bf16(qf0, kf0, s0.v);
  s0.v = wmma_bf16(qf1, kf1, s0.v);

  unsigned short* Pw = Pst[h];
#pragma unroll
  for (int r = 0; r < 8; ++r) {
    float a = s0.f[r] * 0.125f;
    float mx = a;
#pragma unroll
    for (int d = 8; d >= 1; d >>= 1) mx = fmaxf(mx, __shfl_xor(mx, d));
    float p = __expf(a - mx);
    float sum = p;
#pragma unroll
    for (int d = 8; d >= 1; d >>= 1) sum += __shfl_xor(sum, d);
    p /= sum;
    Pw[(rowoff + r) * 32 + coloff]      = f2bf(p);
    Pw[(rowoff + r) * 32 + 16 + coloff] = 0;     // zero-pad K 16 -> 32
  }
  Frag16 pf = frag_ld(Pw, 32);

  AccF o[4];
#pragma unroll
  for (int t = 0; t < 4; ++t)
#pragma unroll
    for (int r = 0; r < 8; ++r) o[t].f[r] = 0.0f;
#pragma unroll
  for (int t = 0; t < 4; ++t) {
    Frag16 vf;
    const unsigned short* p = &Vt[(hc + t * 16 + coloff) * NVm + ((lane & 16) ? 8 : 0)];
    *(uint4*)&vf.u[0] = *(const uint4*)p;
    vf.u[4] = vf.u[5] = vf.u[6] = vf.u[7] = 0u;  // k = 16..31 padded
    o[t].v = wmma_bf16(pf, vf, o[t].v);
  }
#pragma unroll
  for (int r = 0; r < 8; ++r) {
    const long row = row0 + (long)(rowoff + r) * Sm;
#pragma unroll
    for (int t = 0; t < 4; ++t)
      O[row * Dm + hc + t * 16 + coloff] = f2bf(o[t].f[r]);
  }
}

// ---------------- residual + LayerNorm, writes fp32 master + bf16 copy ----------------
__global__ __launch_bounds__(256) void add_ln_k(
    const float* __restrict__ res, const float* __restrict__ y,
    const float* __restrict__ g, const float* __restrict__ bb,
    float* __restrict__ outf, unsigned short* __restrict__ outh) {
  const long row = blockIdx.x;
  const int t = threadIdx.x;
  const float* r = res + row * Dm;
  const float* z = y + row * Dm;
  float v0 = r[t] + z[t];
  float v1 = r[t + 256] + z[t + 256];
  float s = v0 + v1, ss = v0 * v0 + v1 * v1;
#pragma unroll
  for (int d = 16; d >= 1; d >>= 1) { s += __shfl_xor(s, d); ss += __shfl_xor(ss, d); }
  __shared__ float s1[8], s2[8];
  if ((t & 31) == 0) { s1[t >> 5] = s; s2[t >> 5] = ss; }
  __syncthreads();
  float ts = 0.0f, tss = 0.0f;
#pragma unroll
  for (int i = 0; i < 8; ++i) { ts += s1[i]; tss += s2[i]; }
  const float mu = ts * (1.0f / Dm);
  const float var = tss * (1.0f / Dm) - mu * mu;
  const float inv = rsqrtf(var + 1e-5f);
  const float o0 = (v0 - mu) * inv * g[t]       + bb[t];
  const float o1 = (v1 - mu) * inv * g[t + 256] + bb[t + 256];
  outf[row * Dm + t]       = o0;
  outf[row * Dm + t + 256] = o1;
  outh[row * Dm + t]       = f2bf(o0);
  outh[row * Dm + t + 256] = f2bf(o1);
}

// ---------------- host orchestration ----------------
extern "C" void kernel_launch(void* const* d_in, const int* in_sizes, int n_in,
                              void* d_out, int out_size, void* d_ws, size_t ws_size,
                              hipStream_t stream) {
  (void)in_sizes; (void)n_in; (void)out_size; (void)ws_size;
  const float* x     = (const float*)d_in[0];
  const float* sq_wq = (const float*)d_in[1];
  const float* sq_wk = (const float*)d_in[2];
  const float* sq_wv = (const float*)d_in[3];
  const float* sq_wo = (const float*)d_in[4];
  const float* sq_bo = (const float*)d_in[5];
  const float* sq_g  = (const float*)d_in[6];
  const float* sq_b  = (const float*)d_in[7];
  const float* va_wq = (const float*)d_in[8];
  const float* va_wk = (const float*)d_in[9];
  const float* va_wv = (const float*)d_in[10];
  const float* va_wo = (const float*)d_in[11];
  const float* va_bo = (const float*)d_in[12];
  const float* va_g  = (const float*)d_in[13];
  const float* va_b  = (const float*)d_in[14];
  const float* fc1_w = (const float*)d_in[15];
  const float* fc1_b = (const float*)d_in[16];
  const float* fc2_w = (const float*)d_in[17];
  const float* fc2_b = (const float*)d_in[18];
  const float* mlp_g = (const float*)d_in[19];
  const float* mlp_b = (const float*)d_in[20];

  char* p = (char*)d_ws;
  auto carve = [&](size_t bytes) {
    char* r = p; p += (bytes + 255) & ~(size_t)255; return r;
  };
  float*          xw = (float*)carve(Tm * Dm * 4);           // running activation (fp32)
  float*          yb = (float*)carve(Tm * Dm * 4);           // pre-LN GEMM output (fp32)
  unsigned short* xb = (unsigned short*)carve(Tm * Dm * 2);  // bf16 activation
  unsigned short* qb = (unsigned short*)carve(Tm * Dm * 2);
  unsigned short* kb = (unsigned short*)carve(Tm * Dm * 2);
  unsigned short* vb = (unsigned short*)carve(Tm * Dm * 2);
  unsigned short* ab = (unsigned short*)carve(Tm * Dm * 2);  // attention output (bf16)
  unsigned short* w0 = (unsigned short*)carve((size_t)Dm * Dm * 2);
  unsigned short* w1 = (unsigned short*)carve((size_t)Dm * Dm * 2);
  unsigned short* w2 = (unsigned short*)carve((size_t)Dm * Dm * 2);
  unsigned short* w3 = (unsigned short*)carve((size_t)Dm * Dm * 2);
  unsigned short* f1 = (unsigned short*)carve((size_t)DFFm * Dm * 2);
  unsigned short* f2 = (unsigned short*)carve((size_t)Dm * DFFm * 2);
  unsigned short* hb = qb;   // MLP hidden (bf16) aliases Q buffer (attention done by then)

  const long DD = (long)Dm * Dm;
  auto cvt = [&](const float* s, unsigned short* d, long n, int blocks) {
    cvt_bf16_k<<<blocks, 256, 0, stream>>>(s, d, n);
  };
  auto gemm_b16 = [&](const unsigned short* X, const unsigned short* W,
                      unsigned short* Y, int M, int N, int K) {
    gemm_bf16_k<0><<<dim3(M / 256, N / 64), 256, 0, stream>>>(
        X, W, nullptr, nullptr, Y, M, N, K);
  };

  // ---- stage 0: bf16 conversions ----
  cvt(x, xb, Tm * Dm, 4096);
  cvt(sq_wq, w0, DD, 256); cvt(sq_wk, w1, DD, 256);
  cvt(sq_wv, w2, DD, 256); cvt(sq_wo, w3, DD, 256);

  // ---- stage 1: sequence attention over S ----
  gemm_b16(xb, w0, qb, (int)Tm, Dm, Dm);
  gemm_b16(xb, w1, kb, (int)Tm, Dm, Dm);
  gemm_b16(xb, w2, vb, (int)Tm, Dm, Dm);
  attn_seq_k<<<dim3(Bm * NVm, Hm, Sm / 128), 256, 0, stream>>>(qb, kb, vb, ab);
  gemm_bf16_k<3><<<dim3((int)Tm / 256, Dm / 64), 256, 0, stream>>>(
      ab, w3, sq_bo, yb, nullptr, (int)Tm, Dm, Dm);
  add_ln_k<<<(int)Tm, 256, 0, stream>>>(x, yb, sq_g, sq_b, xw, xb);

  // ---- stage 2: variable attention over NV ----
  cvt(va_wq, w0, DD, 256); cvt(va_wk, w1, DD, 256);
  cvt(va_wv, w2, DD, 256); cvt(va_wo, w3, DD, 256);
  gemm_b16(xb, w0, qb, (int)Tm, Dm, Dm);
  gemm_b16(xb, w1, kb, (int)Tm, Dm, Dm);
  gemm_b16(xb, w2, vb, (int)Tm, Dm, Dm);
  attn_var_k<<<dim3(Bm * Sm), 256, 0, stream>>>(qb, kb, vb, ab);
  gemm_bf16_k<3><<<dim3((int)Tm / 256, Dm / 64), 256, 0, stream>>>(
      ab, w3, va_bo, yb, nullptr, (int)Tm, Dm, Dm);
  add_ln_k<<<(int)Tm, 256, 0, stream>>>(xw, yb, va_g, va_b, xw, xb);

  // ---- stage 3: MLP (chunked over tokens so hidden fits workspace) ----
  cvt(fc1_w, f1, (long)DFFm * Dm, 1024);
  cvt(fc2_w, f2, (long)Dm * DFFm, 1024);
  const int CH = 8192;
  for (int c = 0; c < (int)(Tm / CH); ++c) {
    const unsigned short* xc = xb + (size_t)c * CH * Dm;
    gemm_bf16_k<2><<<dim3(CH / 256, DFFm / 64), 256, 0, stream>>>(
        xc, f1, fc1_b, nullptr, hb, CH, DFFm, Dm);
    gemm_bf16_k<3><<<dim3(CH / 256, Dm / 64), 256, 0, stream>>>(
        hb, f2, fc2_b, yb + (size_t)c * CH * Dm, nullptr, CH, Dm, DFFm);
  }
  add_ln_k<<<(int)Tm, 256, 0, stream>>>(xw, yb, mlp_g, mlp_b, (float*)d_out, xb);
}
